// MyModel_52518860096249
// MI455X (gfx1250) — matrix operand order repaired and checked by reading, not compiled, not run on hardware
//
#include <hip/hip_runtime.h>

// ---------------------------------------------------------------------------
// Flash-attention fwd + dropout mask, CDNA5 (gfx1250).
// Pass 1: convert K -> f16 [b][k][d], V -> f16 transposed [b][dv][k] (d_ws).
// Pass 2: flash attention; K/V tiles DMA'd into LDS by the Tensor Data Mover
//         (tensor_load_to_lds, double buffered, s_wait_tensorcnt), softmax
//         row reductions via DPP row_ror (no ds_bpermute), 8 WMMAs / 32 keys.
// Roofline: 268MB drop-mask stream ~11.5us @ 23.3TB/s is the floor; WMMA
// compute ~17.2TFLOP on the f16 path is far faster, so the loop is tuned to
// minimize per-iteration VALU and keep the mask loads streaming.
// ---------------------------------------------------------------------------

typedef __attribute__((ext_vector_type(16))) _Float16 v16h;
typedef __attribute__((ext_vector_type(8)))  _Float16 v8h;
typedef __attribute__((ext_vector_type(4)))  _Float16 v4h;
typedef __attribute__((ext_vector_type(8)))  float    v8f;
typedef __attribute__((ext_vector_type(4)))  float    v4f;
typedef __attribute__((ext_vector_type(4)))  unsigned int u32x4;
typedef __attribute__((ext_vector_type(8)))  int      i32x8;
typedef __attribute__((ext_vector_type(4)))  int      i32x4;

#define B_   16
#define SQ_  2048
#define SK_  2048
#define D_   64
#define DV_  64

#define KB      32            // keys per main-loop iteration
#define QTILE   64            // q rows per block (16 per wave, 4 waves)

// LDS layout (halves). TDM pad fields produce the padded strides directly.
#define VT_STRIDE 40          // 32 + 4DW pad  -> bank-conflict-free, 16B-aligned
#define K_STRIDE  72          // 64 + 4DW pad
#define P_STRIDE  40
#define VT_HALF(i) ((i) * 2560)            // 64*40 per buffer
#define K_HALF(i)  (5120 + (i) * 2304)     // 32*72 per buffer
#define P_HALF     9728                    // 4 waves * 16*40
#define SMEM_HALVES 12288
#define VT_BYTE(i) ((i) * 5120)
#define K_BYTE(i)  (10240 + (i) * 4608)

// ---- DPP row_ror reductions over the 16-lane half-wave --------------------
template <int CTRL>
__device__ __forceinline__ float dpp_mov(float x) {
    return __int_as_float(__builtin_amdgcn_update_dpp(
        0, __float_as_int(x), CTRL, 0xf, 0xf, true));
}
__device__ __forceinline__ float rowmax16(float x) {
    x = fmaxf(x, dpp_mov<0x121>(x));   // row_ror:1
    x = fmaxf(x, dpp_mov<0x122>(x));   // row_ror:2
    x = fmaxf(x, dpp_mov<0x124>(x));   // row_ror:4
    x = fmaxf(x, dpp_mov<0x128>(x));   // row_ror:8
    return x;
}
__device__ __forceinline__ float rowsum16(float x) {
    x += dpp_mov<0x121>(x);
    x += dpp_mov<0x122>(x);
    x += dpp_mov<0x124>(x);
    x += dpp_mov<0x128>(x);
    return x;
}

// ---- Tensor Data Mover: 2D f16 tile -> LDS with row padding ---------------
// D# group0: count=1 | lds_addr | global_addr | type=2
// D# group1: data_size=2B, pad_enable, pad_interval/amount, dims/tile/stride
__device__ __forceinline__ void tdm_load_2d_f16(
    unsigned lds_byte_off, const _Float16* gsrc,
    unsigned tile_w, unsigned tile_h, unsigned long long row_stride_elts,
    unsigned pad_interval_code, unsigned pad_amount_code)
{
    unsigned long long ga = (unsigned long long)(const void*)gsrc;
    u32x4 g0;
    g0.x = 1u;                                            // count=1
    g0.y = lds_byte_off;                                  // lds_addr
    g0.z = (unsigned)ga;                                  // global_addr[31:0]
    g0.w = (unsigned)((ga >> 32) & 0x01ffffffu) | (2u << 30);  // [56:32]|type=2

    i32x8 g1;
    g1[0] = (int)((1u << 16)                 // data_size = 2 bytes
                | (1u << 20)                 // pad_enable
                | (pad_interval_code << 22)
                | (pad_amount_code  << 25));
    g1[1] = (int)((tile_w & 0xffffu) << 16);              // tensor_dim0 lo16
    g1[2] = (int)((tile_h & 0xffffu) << 16);              // tensor_dim1 lo16
    g1[3] = (int)((tile_w & 0xffffu) << 16);              // tile_dim0
    g1[4] = (int)(tile_h & 0xffffu);                      // tile_dim1 (dim2=0)
    g1[5] = (int)(unsigned)(row_stride_elts & 0xffffffffull);   // dim0_stride lo
    g1[6] = (int)(unsigned)((row_stride_elts >> 32) & 0xffffull);
    g1[7] = 0;

    i32x4 z4 = (i32x4){0, 0, 0, 0};                       // groups 2/3 unused (2D)
    i32x8 z8 = (i32x8){0, 0, 0, 0, 0, 0, 0, 0};
    __builtin_amdgcn_tensor_load_to_lds(g0, g1, z4, z4, z8, 0);
}

// ---------------------------------------------------------------------------
// Pass 1: K f32 -> f16 (same layout); V f32 -> f16 transposed [b][dv][k].
// ---------------------------------------------------------------------------
__global__ __launch_bounds__(256)
void prep_f16(const float* __restrict__ K, const float* __restrict__ V,
              _Float16* __restrict__ Kh, _Float16* __restrict__ VTh)
{
    const size_t t  = (size_t)blockIdx.x * 256 + threadIdx.x;
    const size_t i0 = t * 4;                       // element index (mult of 4)

    const v4f kv = *(const v4f*)(K + i0);
    v4h kh;
    kh[0] = (_Float16)kv[0]; kh[1] = (_Float16)kv[1];
    kh[2] = (_Float16)kv[2]; kh[3] = (_Float16)kv[3];
    *(v4h*)(Kh + i0) = kh;

    const v4f vv = *(const v4f*)(V + i0);
    const size_t dv0 = i0 & (DV_ - 1);             // dv0..dv0+3 within one row
    const size_t bk  = i0 >> 6;                    // b*SK + k
    const size_t k   = bk & (SK_ - 1);
    const size_t b   = bk >> 11;
    _Float16* dst = VTh + (b * DV_ + dv0) * SK_ + k;
    dst[0 * SK_] = (_Float16)vv[0];
    dst[1 * SK_] = (_Float16)vv[1];
    dst[2 * SK_] = (_Float16)vv[2];
    dst[3 * SK_] = (_Float16)vv[3];
}

// ---------------------------------------------------------------------------
// Pass 2: flash attention with TDM-staged tiles.
// ---------------------------------------------------------------------------
__global__ __launch_bounds__(128)
void fa_fwd_wmma(const float* __restrict__ Q, const _Float16* __restrict__ Kh,
                 const _Float16* __restrict__ VTh, const float* __restrict__ M,
                 float* __restrict__ O)
{
    __shared__ _Float16 smem[SMEM_HALVES];

    const int lane = threadIdx.x & 31;
    const int wave = threadIdx.x >> 5;
    const int hi   = lane >> 4;
    const int ln   = lane & 15;

    const int b      = blockIdx.y;
    const int q_base = blockIdx.x * QTILE + wave * 16;

    const float*    Qb = Q   + (size_t)b * SQ_ * D_;
    const _Float16* Kb = Kh  + (size_t)b * SK_ * D_;
    const _Float16* Vb = VTh + (size_t)b * DV_ * SK_;
    const float*    Mb = M   + (size_t)b * SQ_ * (size_t)SK_;
    float*          Ob = O   + (size_t)b * SQ_ * DV_;

    const float kS2      = 0.125f * 1.44269504f;   // 1/sqrt(D) * log2(e)
    const float kDropP   = 0.1f;
    const float kInvKeep = 1.0f / 0.9f;

    // ---- Q A-fragments (16x32 f16), two K-chunks over D=64 ----------------
    v16h qa[2];
    {
        const float* qrow = Qb + (size_t)(q_base + ln) * D_;
        #pragma unroll
        for (int f = 0; f < 2; ++f) {
            const int k0 = f * 32 + hi * 16;
            const int k1 = f * 32 + hi * 16 + 8;
            #pragma unroll
            for (int e = 0; e < 8; ++e) {
                qa[f][e]     = (_Float16)qrow[k0 + e];
                qa[f][e + 8] = (_Float16)qrow[k1 + e];
            }
        }
    }

    float m_run[8], l_run[8];
    v8f   oacc[4];
    #pragma unroll
    for (int j = 0; j < 8; ++j) { m_run[j] = -1e30f; l_run[j] = 0.0f; }
    #pragma unroll
    for (int t = 0; t < 4; ++t) oacc[t] = (v8f){};

    // ---- pre-issue TDM for tile 0 (double buffered) -----------------------
    if (wave == 0) {
        tdm_load_2d_f16(VT_BYTE(0), Vb, 32, 64, SK_, 3, 3);          // VT tile
        tdm_load_2d_f16(K_BYTE(0),  Kb, 64, 32, D_,  4, 3);          // K tile
    }

    const int nIter = SK_ / KB;                    // 64
    for (int it = 0; it < nIter; ++it) {
        const int kb  = it * KB;
        const int cur = it & 1;

        if (wave == 0) __builtin_amdgcn_s_wait_tensorcnt(0);
        __syncthreads();                           // tile(it) visible, WAR safe

        if (wave == 0 && it + 1 < nIter) {         // DMA next tile during compute
            const int nkb = kb + KB;
            tdm_load_2d_f16(VT_BYTE(cur ^ 1), Vb + nkb, 32, 64, SK_, 3, 3);
            tdm_load_2d_f16(K_BYTE(cur ^ 1), Kb + (size_t)nkb * D_, 64, 32, D_, 4, 3);
        }

        const _Float16* Kl = &smem[K_HALF(cur)];
        const _Float16* Vl = &smem[VT_HALF(cur)];

        // ---- S = Q K^T, two 16-key subtiles, K=64 split 2x32 --------------
        v8f s[2];
        #pragma unroll
        for (int T = 0; T < 2; ++T) {
            const _Float16* kr = Kl + (T * 16 + ln) * K_STRIDE;
            v8h a0 = *(const v8h*)(kr + hi * 16);
            v8h a1 = *(const v8h*)(kr + hi * 16 + 8);
            v8h b0 = *(const v8h*)(kr + 32 + hi * 16);
            v8h b1 = *(const v8h*)(kr + 32 + hi * 16 + 8);
            v16h kf0 = __builtin_shufflevector(a0, a1, 0,1,2,3,4,5,6,7,8,9,10,11,12,13,14,15);
            v16h kf1 = __builtin_shufflevector(b0, b1, 0,1,2,3,4,5,6,7,8,9,10,11,12,13,14,15);
            v8f acc = (v8f){};
            acc = __builtin_amdgcn_wmma_f32_16x16x32_f16(false, qa[0], false, kf0,
                                                         (short)0, acc, false, false);
            acc = __builtin_amdgcn_wmma_f32_16x16x32_f16(false, qa[1], false, kf1,
                                                         (short)0, acc, false, false);
            s[T] = acc;
        }

        // ---- online softmax in exp2 domain, DPP row reductions ------------
        #pragma unroll
        for (int j = 0; j < 8; ++j) {
            const float v0 = s[0][j] * kS2;
            const float v1 = s[1][j] * kS2;
            const float mx   = rowmax16(fmaxf(v0, v1));
            const float mnew = fmaxf(m_run[j], mx);
            const float alpha = __builtin_amdgcn_exp2f(m_run[j] - mnew);
            const float p0 = __builtin_amdgcn_exp2f(v0 - mnew);
            const float p1 = __builtin_amdgcn_exp2f(v1 - mnew);
            const float rs = rowsum16(p0 + p1);    // pre-dropout row sum
            l_run[j] = l_run[j] * alpha + rs;
            m_run[j] = mnew;
            s[0][j] = p0;
            s[1][j] = p1;
            oacc[0][j] *= alpha; oacc[1][j] *= alpha;
            oacc[2][j] *= alpha; oacc[3][j] *= alpha;
        }

        // ---- dropout (the HBM-critical 268MB stream) + P -> LDS -----------
        #pragma unroll
        for (int T = 0; T < 2; ++T) {
            #pragma unroll
            for (int j = 0; j < 8; ++j) {
                const int   m  = j + 8 * hi;
                const float dm = Mb[(size_t)(q_base + m) * SK_ + kb + T * 16 + ln];
                const float pv = (dm > kDropP) ? s[T][j] : 0.0f;
                smem[P_HALF + (wave * 16 + m) * P_STRIDE + T * 16 + ln] = (_Float16)pv;
            }
        }
        __syncthreads();                           // P visible (cross-lane)

        // ---- O += P V : P as 16x32 A-fragment, VT rows as B-fragments -----
        const _Float16* prow = &smem[P_HALF + (wave * 16 + ln) * P_STRIDE];
        v8h p0 = *(const v8h*)(prow + hi * 16);
        v8h p1 = *(const v8h*)(prow + hi * 16 + 8);
        v16h pa = __builtin_shufflevector(p0, p1, 0,1,2,3,4,5,6,7,8,9,10,11,12,13,14,15);
        #pragma unroll
        for (int t = 0; t < 4; ++t) {
            const _Float16* vrow = Vl + (t * 16 + ln) * VT_STRIDE + hi * 16;
            v8h b0 = *(const v8h*)(vrow);
            v8h b1 = *(const v8h*)(vrow + 8);
            v16h vb = __builtin_shufflevector(b0, b1, 0,1,2,3,4,5,6,7,8,9,10,11,12,13,14,15);
            oacc[t] = __builtin_amdgcn_wmma_f32_16x16x32_f16(false, pa, false, vb,
                                                             (short)0, oacc[t], false, false);
        }
    }

    // ---- epilogue: O = O / l * 1/(1-p) ------------------------------------
    #pragma unroll
    for (int j = 0; j < 8; ++j) {
        const float inv = kInvKeep / l_run[j];
        float* orow = Ob + (size_t)(q_base + j + 8 * hi) * DV_;
        #pragma unroll
        for (int t = 0; t < 4; ++t)
            orow[t * 16 + ln] = oacc[t][j] * inv;
    }
}

extern "C" void kernel_launch(void* const* d_in, const int* in_sizes, int n_in,
                              void* d_out, int out_size, void* d_ws, size_t ws_size,
                              hipStream_t stream) {
    const float* Q = (const float*)d_in[0];
    const float* K = (const float*)d_in[1];
    const float* V = (const float*)d_in[2];
    const float* M = (const float*)d_in[3];
    float*       O = (float*)d_out;

    _Float16* Kh  = (_Float16*)d_ws;                       // 4 MB
    _Float16* VTh = Kh + (size_t)B_ * SK_ * D_;            // 4 MB (needs 8MB ws)

    const size_t elems = (size_t)B_ * SK_ * D_;            // 2M, 4 per thread
    prep_f16<<<dim3((unsigned)(elems / (4 * 256))), dim3(256), 0, stream>>>(K, V, Kh, VTh);

    fa_fwd_wmma<<<dim3(SQ_ / QTILE, B_), dim3(128), 0, stream>>>(Q, Kh, VTh, M, O);
}
